// Former2Mobile_11450382811319
// MI455X (gfx1250) — matrix operand order repaired and checked by reading, not compiled, not run on hardware
//
#include <hip/hip_runtime.h>
#include <hip/hip_bf16.h>

// ---------- types ----------
typedef __bf16 bf16_t;
typedef __bf16 v16bf __attribute__((ext_vector_type(16)));
typedef __bf16 v8bf  __attribute__((ext_vector_type(8)));
typedef float  v8f   __attribute__((ext_vector_type(8)));

// ---------- geometry ----------
constexpr int B_  = 16;
constexpr int C_  = 512;
constexpr int HW_ = 4096;
constexpr int M_  = 64;
constexpr int D_  = 512;
constexpr int NH  = 8;
constexpr int HD  = 64;
constexpr int QT  = 64;          // queries per block
constexpr int LDA = D_ + 8;      // bf16 LDS row stride (520): keeps 16B alignment
constexpr int LDS_S = 68;        // f32 score stride
constexpr int LDS_P = 72;        // bf16 prob stride

// LDS overlay offsets (bytes) for the attention kernel
constexpr int SM_A = 0;                      // [64][520] bf16 LN(spatial) -> A, later O
constexpr int SM_Q = 66560;                  // [64][520] bf16 Q
constexpr int SM_S = 133120;                 // [64][68]  f32 scores
constexpr int SM_P = 150528;                 // [64][72]  bf16 probs
constexpr int SM_F = 66560;                  // [512][64] f32 out tile (reuses Q)
constexpr size_t SMEM_ATTN = 197632;         // peak
constexpr size_t SMEM_KV   = (size_t)M_ * LDA * sizeof(bf16_t); // 66560

// ---------- helpers ----------
__device__ inline bf16_t f2bf(float f) {
  unsigned u = __builtin_bit_cast(unsigned, f);
  unsigned r = u + 0x7FFFu + ((u >> 16) & 1u);       // round-to-nearest-even
  unsigned short h = (unsigned short)(r >> 16);
  return __builtin_bit_cast(bf16_t, h);
}
__device__ inline float bf2f(bf16_t b) {
  unsigned short h = __builtin_bit_cast(unsigned short, b);
  unsigned u = ((unsigned)h) << 16;
  return __builtin_bit_cast(float, u);
}
__device__ inline float wave_sum(float v) {
  #pragma unroll
  for (int off = 16; off > 0; off >>= 1) v += __shfl_xor(v, off, 32);
  return v;
}
__device__ inline float wave_max(float v) {
  #pragma unroll
  for (int off = 16; off > 0; off >>= 1) v = fmaxf(v, __shfl_xor(v, off, 32));
  return v;
}
__device__ inline v8f vzero() {
  v8f v;
  #pragma unroll
  for (int j = 0; j < 8; ++j) v[j] = 0.f;
  return v;
}
__device__ inline v8f wmma_bf16(v16bf a, v16bf b, v8f c) {
  // D = A(16x32 bf16) * B(32x16 bf16) + C(16x16 f32)
  return __builtin_amdgcn_wmma_f32_16x16x32_bf16(false, a, false, b, (short)0, c,
                                                 false, false);
}
// A fragment: row-major [m][k], lane L<16: M=L, K=k0..k0+15;
// lanes 16-31: same M, K=k0+16..k0+31 (matches B fragment order)
__device__ inline v16bf load_frag_a(const bf16_t* base, int lda, int m0, int k0, int lane) {
  int m  = m0 + (lane & 15);
  int ks = k0 + ((lane & 16) ? 16 : 0);
  const bf16_t* p = base + m * lda + ks;
  v8bf lo = *(const v8bf*)p;
  v8bf hi = *(const v8bf*)(p + 8);
  return __builtin_shufflevector(lo, hi, 0,1,2,3,4,5,6,7,8,9,10,11,12,13,14,15);
}
// B fragment from transposed weight [n][k]: lane L<16: N=L, K=k0..k0+15;
// lanes 16-31: N=L-16, K=k0+16..k0+31 (16 contiguous bf16 = one 32B load)
__device__ inline v16bf load_frag_b(const bf16_t* base, int ldb, int n0, int k0, int lane) {
  int n  = n0 + (lane & 15);
  int ks = k0 + ((lane & 16) ? 16 : 0);
  return *(const v16bf*)(base + n * ldb + ks);
}

// ---------- kernel 0: f32 weight -> bf16 transposed [n][k] ----------
__global__ void transpose_to_bf16(const float* __restrict__ W, bf16_t* __restrict__ Wt,
                                  int K, int N) {
  int idx = blockIdx.x * blockDim.x + threadIdx.x;
  if (idx >= K * N) return;
  int n = idx / K;
  int k = idx - n * K;
  Wt[idx] = f2bf(W[(size_t)k * N + n]);
}

// ---------- kernel 1: LN(tokens) @ Wkv -> K (row major) and V^T ----------
__global__ __launch_bounds__(256, 1)
void kv_kernel(const float* __restrict__ tokens, const bf16_t* __restrict__ Wkvt,
               const float* __restrict__ bkv, const float* __restrict__ g_tk,
               const float* __restrict__ b_tk, bf16_t* __restrict__ Kd,
               bf16_t* __restrict__ Vt) {
  extern __shared__ char smem[];
  bf16_t* A = (bf16_t*)smem;                 // [64][LDA] LN(tokens) bf16
  const int b = blockIdx.y;
  const int nchunk = blockIdx.x;             // 256 output cols each
  const int tid = threadIdx.x, lane = tid & 31, wave = tid >> 5;

  // LayerNorm: wave w -> token rows 8w..8w+7, lane holds 16 contiguous channels
  for (int m = wave * 8; m < wave * 8 + 8; ++m) {
    const float* row = tokens + ((size_t)b * M_ + m) * D_;
    float x[16], s = 0.f, ss = 0.f;
    #pragma unroll
    for (int i = 0; i < 16; ++i) { x[i] = row[lane * 16 + i]; s += x[i]; ss += x[i] * x[i]; }
    s = wave_sum(s); ss = wave_sum(ss);
    float mu = s * (1.f / 512.f);
    float rs = rsqrtf(ss * (1.f / 512.f) - mu * mu + 1e-5f);
    #pragma unroll
    for (int i = 0; i < 16; ++i) {
      int c = lane * 16 + i;
      A[m * LDA + c] = f2bf((x[i] - mu) * rs * g_tk[c] + b_tk[c]);
    }
  }
  __syncthreads();

  // GEMM (64x512) @ Wkvt-chunk -> 64x256
  const int mt = wave & 3;
  const int ntb = (wave >> 2) * 8;
  v8f acc[8];
  #pragma unroll
  for (int j = 0; j < 8; ++j) acc[j] = vzero();
  for (int ks = 0; ks < D_; ks += 32) {
    v16bf af = load_frag_a(A, LDA, mt * 16, ks, lane);
    #pragma unroll
    for (int j = 0; j < 8; ++j) {
      int ncol = nchunk * 256 + (ntb + j) * 16;
      v16bf bf_ = load_frag_b(Wkvt, D_, ncol, ks, lane);
      acc[j] = wmma_bf16(af, bf_, acc[j]);
    }
  }
  #pragma unroll
  for (int j = 0; j < 8; ++j) {
    int n = nchunk * 256 + (ntb + j) * 16 + (lane & 15);
    float bias = bkv[n];
    #pragma unroll
    for (int r = 0; r < 8; ++r) {
      int m = mt * 16 + r + ((lane & 16) ? 8 : 0);
      float v = acc[j][r] + bias;
      if (n < D_) Kd[((size_t)b * M_ + m) * D_ + n] = f2bf(v);          // K[b][m][d]
      else        Vt[((size_t)b * D_ + (n - D_)) * M_ + m] = f2bf(v);   // V^T[b][d][m]
    }
  }
}

// ---------- kernel 2: fused LN + Qproj + attention + outproj + residual ----------
__global__ __launch_bounds__(256, 1)
void attn_kernel(const float* __restrict__ spatial, const bf16_t* __restrict__ Wqt,
                 const float* __restrict__ bq, const bf16_t* __restrict__ Woutt,
                 const float* __restrict__ bout, const float* __restrict__ g_sp,
                 const float* __restrict__ b_sp, const bf16_t* __restrict__ Kd,
                 const bf16_t* __restrict__ Vt, float* __restrict__ out) {
  extern __shared__ char smem[];
  bf16_t* A = (bf16_t*)(smem + SM_A);   // LN(spatial) tile, later reused as O
  bf16_t* Q = (bf16_t*)(smem + SM_Q);
  float*  S = (float*)(smem + SM_S);
  bf16_t* P = (bf16_t*)(smem + SM_P);
  bf16_t* O = (bf16_t*)(smem + SM_A);   // attention output (A is dead by then)
  float*  F = (float*)(smem + SM_F);    // final f32 tile [c][q] (Q/S/P dead)

  const int b  = blockIdx.y;
  const int q0 = blockIdx.x * QT;
  const int tid = threadIdx.x, lane = tid & 31, wave = tid >> 5;

  // --- P1: transposing load spatial[b][c][q0..q0+63] -> A[q][c] (bf16) ---
  for (int idx = tid; idx < C_ * (QT / 4); idx += 256) {
    int c  = idx >> 4;
    int q4 = (idx & 15) * 4;
    const float4 v = *(const float4*)(spatial + ((size_t)b * C_ + c) * HW_ + q0 + q4);
    A[(q4 + 0) * LDA + c] = f2bf(v.x);
    A[(q4 + 1) * LDA + c] = f2bf(v.y);
    A[(q4 + 2) * LDA + c] = f2bf(v.z);
    A[(q4 + 3) * LDA + c] = f2bf(v.w);
  }
  __syncthreads();

  // --- P2: per-query LayerNorm in LDS ---
  for (int q = wave * 8; q < wave * 8 + 8; ++q) {
    float x[16], s = 0.f, ss = 0.f;
    #pragma unroll
    for (int i = 0; i < 16; ++i) {
      x[i] = bf2f(A[q * LDA + lane * 16 + i]);
      s += x[i]; ss += x[i] * x[i];
    }
    s = wave_sum(s); ss = wave_sum(ss);
    float mu = s * (1.f / 512.f);
    float rs = rsqrtf(ss * (1.f / 512.f) - mu * mu + 1e-5f);
    #pragma unroll
    for (int i = 0; i < 16; ++i) {
      int c = lane * 16 + i;
      A[q * LDA + c] = f2bf((x[i] - mu) * rs * g_sp[c] + b_sp[c]);
    }
  }
  __syncthreads();

  // --- P3: Q = A @ Wq^T(stored [n][k]) + bq ---
  {
    const int mt = wave & 3;
    for (int g = 0; g < 2; ++g) {
      const int ntb = (wave >> 2) * 16 + g * 8;
      v8f acc[8];
      #pragma unroll
      for (int j = 0; j < 8; ++j) acc[j] = vzero();
      for (int ks = 0; ks < D_; ks += 32) {
        __builtin_prefetch(Wqt + (size_t)(ntb * 16) * D_ + ks + 128, 0, 1);
        v16bf af = load_frag_a(A, LDA, mt * 16, ks, lane);
        #pragma unroll
        for (int j = 0; j < 8; ++j) {
          v16bf bf_ = load_frag_b(Wqt, D_, (ntb + j) * 16, ks, lane);
          acc[j] = wmma_bf16(af, bf_, acc[j]);
        }
      }
      #pragma unroll
      for (int j = 0; j < 8; ++j) {
        int n = (ntb + j) * 16 + (lane & 15);
        float bias = bq[n];
        #pragma unroll
        for (int r = 0; r < 8; ++r) {
          int m = mt * 16 + r + ((lane & 16) ? 8 : 0);
          Q[m * LDA + n] = f2bf(acc[j][r] + bias);
        }
      }
    }
  }
  __syncthreads();

  // --- P4: per-head attention (K, V^T streamed from L2-resident bf16) ---
  const float scale = 0.125f;  // 1/sqrt(64)
  for (int h = 0; h < NH; ++h) {
    {  // scores S = Q_h @ K_h^T * scale
      const int mt = wave & 3;
      const int ntb = (wave >> 2) * 2;
      v8f acc[2]; acc[0] = vzero(); acc[1] = vzero();
      for (int ks = 0; ks < HD; ks += 32) {
        v16bf af = load_frag_a(Q, LDA, mt * 16, h * HD + ks, lane);
        #pragma unroll
        for (int j = 0; j < 2; ++j) {
          int t  = (ntb + j) * 16 + (lane & 15);
          int kk = h * HD + ks + ((lane & 16) ? 16 : 0);
          v16bf bf_ = *(const v16bf*)(Kd + ((size_t)b * M_ + t) * D_ + kk);
          acc[j] = wmma_bf16(af, bf_, acc[j]);
        }
      }
      #pragma unroll
      for (int j = 0; j < 2; ++j) {
        int n = (ntb + j) * 16 + (lane & 15);
        #pragma unroll
        for (int r = 0; r < 8; ++r) {
          int m = mt * 16 + r + ((lane & 16) ? 8 : 0);
          S[m * LDS_S + n] = acc[j][r] * scale;
        }
      }
    }
    __syncthreads();
    // softmax over 64 tokens; lane handles 2 columns
    for (int q = wave * 8; q < wave * 8 + 8; ++q) {
      float v0 = S[q * LDS_S + lane];
      float v1 = S[q * LDS_S + 32 + lane];
      float mx = wave_max(fmaxf(v0, v1));
      float e0 = __expf(v0 - mx), e1 = __expf(v1 - mx);
      float inv = 1.f / wave_sum(e0 + e1);
      P[q * LDS_P + lane]      = f2bf(e0 * inv);
      P[q * LDS_P + 32 + lane] = f2bf(e1 * inv);
    }
    __syncthreads();
    {  // O_h = P @ V_h  (V stored transposed: contiguous in tokens)
      const int mt = wave & 3;
      const int ntb = (wave >> 2) * 2;
      v8f acc[2]; acc[0] = vzero(); acc[1] = vzero();
      for (int ks = 0; ks < M_; ks += 32) {
        v16bf af = load_frag_a(P, LDS_P, mt * 16, ks, lane);
        #pragma unroll
        for (int j = 0; j < 2; ++j) {
          int n  = (ntb + j) * 16 + (lane & 15);
          int kk = ks + ((lane & 16) ? 16 : 0);
          v16bf bf_ = *(const v16bf*)(Vt + ((size_t)b * D_ + h * HD + n) * M_ + kk);
          acc[j] = wmma_bf16(af, bf_, acc[j]);
        }
      }
      #pragma unroll
      for (int j = 0; j < 2; ++j) {
        int n = (ntb + j) * 16 + (lane & 15);
        #pragma unroll
        for (int r = 0; r < 8; ++r) {
          int m = mt * 16 + r + ((lane & 16) ? 8 : 0);
          O[m * LDA + h * HD + n] = f2bf(acc[j][r]);
        }
      }
    }
    __syncthreads();
  }

  // --- P5: F = O @ Wout^T + bout, staged [c][q] for coalesced residual store ---
  {
    const int mt = wave & 3;
    for (int g = 0; g < 2; ++g) {
      const int ntb = (wave >> 2) * 16 + g * 8;
      v8f acc[8];
      #pragma unroll
      for (int j = 0; j < 8; ++j) acc[j] = vzero();
      for (int ks = 0; ks < D_; ks += 32) {
        __builtin_prefetch(Woutt + (size_t)(ntb * 16) * D_ + ks + 128, 0, 1);
        v16bf af = load_frag_a(O, LDA, mt * 16, ks, lane);
        #pragma unroll
        for (int j = 0; j < 8; ++j) {
          v16bf bf_ = load_frag_b(Woutt, D_, (ntb + j) * 16, ks, lane);
          acc[j] = wmma_bf16(af, bf_, acc[j]);
        }
      }
      #pragma unroll
      for (int j = 0; j < 8; ++j) {
        int n = (ntb + j) * 16 + (lane & 15);
        float bias = bout[n];
        #pragma unroll
        for (int r = 0; r < 8; ++r) {
          int m = mt * 16 + r + ((lane & 16) ? 8 : 0);
          F[n * QT + m] = acc[j][r] + bias;
        }
      }
    }
  }
  __syncthreads();

  // residual add + coalesced float4 store in (B, C, HW) layout
  for (int idx = tid; idx < C_ * (QT / 4); idx += 256) {
    int c  = idx >> 4;
    int q4 = (idx & 15) * 4;
    size_t g = ((size_t)b * C_ + c) * HW_ + q0 + q4;
    float4 r = *(const float4*)(spatial + g);
    float4 f = *(const float4*)(F + c * QT + q4);
    float4 o;
    o.x = r.x + f.x; o.y = r.y + f.y; o.z = r.z + f.z; o.w = r.w + f.w;
    *(float4*)(out + g) = o;
  }
}

// ---------- host ----------
extern "C" void kernel_launch(void* const* d_in, const int* in_sizes, int n_in,
                              void* d_out, int out_size, void* d_ws, size_t ws_size,
                              hipStream_t stream) {
  (void)in_sizes; (void)n_in; (void)out_size; (void)ws_size;
  const float* spatial = (const float*)d_in[0];
  const float* tokens  = (const float*)d_in[1];
  const float* Wq      = (const float*)d_in[2];
  const float* bq      = (const float*)d_in[3];
  const float* Wkv     = (const float*)d_in[4];
  const float* bkv     = (const float*)d_in[5];
  const float* Wout    = (const float*)d_in[6];
  const float* bout    = (const float*)d_in[7];
  const float* g_sp    = (const float*)d_in[8];
  const float* b_sp    = (const float*)d_in[9];
  const float* g_tk    = (const float*)d_in[10];
  const float* b_tk    = (const float*)d_in[11];
  float* out = (float*)d_out;

  char* ws = (char*)d_ws;
  bf16_t* Wqt   = (bf16_t*)(ws);                                  // 512KB
  bf16_t* Wkvt  = (bf16_t*)(ws + 524288);                         // 1MB
  bf16_t* Woutt = (bf16_t*)(ws + 524288 + 1048576);               // 512KB
  bf16_t* Kd    = (bf16_t*)(ws + 2 * 524288 + 1048576);           // 1MB
  bf16_t* Vt    = (bf16_t*)(ws + 2 * 524288 + 2 * 1048576);       // 1MB

  transpose_to_bf16<<<(512 * 512 + 255) / 256, 256, 0, stream>>>(Wq, Wqt, 512, 512);
  transpose_to_bf16<<<(512 * 1024 + 255) / 256, 256, 0, stream>>>(Wkv, Wkvt, 512, 1024);
  transpose_to_bf16<<<(512 * 512 + 255) / 256, 256, 0, stream>>>(Wout, Woutt, 512, 512);

  dim3 gkv(4, 16);
  kv_kernel<<<gkv, 256, SMEM_KV, stream>>>(tokens, Wkvt, bkv, g_tk, b_tk, Kd, Vt);

  dim3 gat(64, 16);
  attn_kernel<<<gat, 256, SMEM_ATTN, stream>>>(spatial, Wqt, bq, Woutt, bout,
                                               g_sp, b_sp, Kd, Vt, out);
}
